// MultiHeadAttention_26199300505914
// MI455X (gfx1250) — compile-verified, matrix operand order checked by reading, not run on hardware
//
#include <hip/hip_runtime.h>
#include <hip/hip_bf16.h>
#include <math.h>

// ---------------------------------------------------------------------------
// MHA for gfx1250 (CDNA5).  All matmuls via v_wmma_f32_16x16x32_bf16.
// Workgroup-shared operand tiles staged into LDS with the Tensor Data Mover
// (tensor_load_to_lds, TENSORcnt), double-buffered.
// Workspace (~92 MB):
//   Xb  [8192][1024]  bf16   x converted
//   Wqt/Wkt/Wvt [1024][1024] bf16  weights transposed to [out][in]
//   Wot [1024][1024]  bf16   output proj as [d_out][h*64+e]
//   bqc/bkc/bvc/boc [1024]   f32 biases (boc = sum_h bo[h])
//   Qb/Kb [8192][1024] bf16  row-major (col = h*64+e)
//   Vt  [4096][2048]  bf16   V transposed: row = (b*H+h)*64+e, col = s
//   Hd  [8192][1024]  bf16   attention head outputs
// ---------------------------------------------------------------------------

typedef __attribute__((ext_vector_type(16))) __bf16 v16bf;
typedef __attribute__((ext_vector_type(8)))  __bf16 v8bf;
typedef __attribute__((ext_vector_type(8)))  float  v8f;

#define WMMA_BF16(a, b, c) \
  __builtin_amdgcn_wmma_f32_16x16x32_bf16(false, (a), false, (b), (short)0, (c), false, false)

static constexpr int BB = 4, S = 2048, DM = 1024, H = 16, DH = 64;
static constexpr int M  = BB * S;   // 8192 token rows

// ------------------------- Tensor Data Mover glue ---------------------------
#if defined(__HIP_DEVICE_COMPILE__) && __has_builtin(__builtin_amdgcn_tensor_load_to_lds)
#define HAVE_TDM 1
typedef unsigned int u32x4 __attribute__((ext_vector_type(4)));
typedef int          i32x4 __attribute__((ext_vector_type(4)));
typedef int          i32x8 __attribute__((ext_vector_type(8)));

// 2D tile load: tile_d1 rows of tile_d0 contiguous 2-byte elements, source
// row pitch stride0 (elements), packed into LDS at lds_off (row pitch =
// tile_d0 elements).  D# field packing per CDNA5 ISA ch.8 (groups 0/1).
__device__ __forceinline__ void tdm_load_2d(unsigned int lds_off,
                                            const __bf16* gsrc,
                                            unsigned int tensor_d0,
                                            unsigned int tensor_d1,
                                            unsigned int tile_d0,
                                            unsigned int tile_d1,
                                            unsigned int stride0) {
  unsigned long long ga = (unsigned long long)(uintptr_t)gsrc;
  u32x4 g0;
  g0[0] = 1u;                                   // count=1, user descriptor
  g0[1] = lds_off;                              // lds_addr (bytes)
  g0[2] = (unsigned int)ga;                     // global_addr[31:0]
  g0[3] = (unsigned int)((ga >> 32) & 0x01FFFFFFu) | (2u << 30);  // [56:32]|type=2
  i32x8 g1;
  g1[0] = (int)(1u << 16);                      // workgroup_mask=0, data_size=1 (2B)
  g1[1] = (int)((tensor_d0 & 0xFFFFu) << 16);   // tensor_dim0[15:0]
  g1[2] = (int)((tensor_d0 >> 16) | ((tensor_d1 & 0xFFFFu) << 16));
  g1[3] = (int)((tensor_d1 >> 16) | (tile_d0 << 16));   // tile_dim0
  g1[4] = (int)(tile_d1 & 0xFFFFu);             // tile_dim1 (tile_dim2=0)
  g1[5] = (int)stride0;                         // tensor_dim0_stride[31:0]
  g1[6] = 0;                                    // stride[47:32], dim1_stride lo
  g1[7] = 0;
  i32x4 z4 = {0, 0, 0, 0};
#if defined(__clang_major__) && (__clang_major__ >= 23)
  i32x8 z8 = {0, 0, 0, 0, 0, 0, 0, 0};
  __builtin_amdgcn_tensor_load_to_lds(g0, g1, z4, z4, z8, 0);
#else
  __builtin_amdgcn_tensor_load_to_lds(g0, g1, z4, z4, 0);
#endif
}
#else
#define HAVE_TDM 0
#endif

__device__ __forceinline__ unsigned int lds_offset_of(const void* p) {
  // Generic pointers into LDS carry the LDS byte offset in the low 32 bits.
  return (unsigned int)(uintptr_t)p;
}

__device__ __forceinline__ void tensor_wait_barrier(bool issuer) {
#if HAVE_TDM
  if (issuer) __builtin_amdgcn_s_wait_tensorcnt((short)0);
#endif
  __syncthreads();
}

// Load a 16x32 bf16 WMMA fragment (A layout; B 32x16 uses the identical
// per-lane pattern when the source is stored [N][K] row-major).
// Lane L: row = base_row + (L&15); halves [kb, kb+8) and [kb+16, kb+24)
// with kb = k0 + (L>=16 ? 8 : 0).  Two 16-byte loads per lane.
__device__ __forceinline__ v16bf load_frag(const __bf16* __restrict__ base,
                                           int row, int ld, int k0, int lane) {
  const int r  = row + (lane & 15);
  const int kb = k0 + ((lane >> 4) << 3);
  const __bf16* p = base + (size_t)r * ld + kb;
  v8bf lo = *(const v8bf*)(p);
  v8bf hi = *(const v8bf*)(p + 16);
  v16bf f;
#pragma unroll
  for (int i = 0; i < 8; ++i) { f[i] = lo[i]; f[i + 8] = hi[i]; }
  return f;
}

// ----------------------------- prep kernels --------------------------------

__global__ __launch_bounds__(256) void cvt_x_kernel(const float* __restrict__ x,
                                                    __bf16* __restrict__ xb, int n) {
  int i = blockIdx.x * 256 + threadIdx.x;
  if (i < n) xb[i] = (__bf16)x[i];
}

// wq/wk/wv fp32 [H][DM][DH] -> W*t bf16 [DM out][DM in]  (out row = h*64+e)
// wo fp32 [H][DH][DM]       -> Wot bf16 [DM out][DM in]  (in col = h*64+e)
__global__ __launch_bounds__(256) void prep_w_kernel(
    const float* __restrict__ wq, const float* __restrict__ wk,
    const float* __restrict__ wv, const float* __restrict__ wo,
    __bf16* __restrict__ Wqt, __bf16* __restrict__ Wkt,
    __bf16* __restrict__ Wvt, __bf16* __restrict__ Wot) {
  int i = blockIdx.x * 256 + threadIdx.x;   // over 1024*1024
  int r = i >> 10, c = i & 1023;
  int src = (r >> 6) * (DM * DH) + c * DH + (r & 63);
  Wqt[i] = (__bf16)wq[src];
  Wkt[i] = (__bf16)wk[src];
  Wvt[i] = (__bf16)wv[src];
  Wot[i] = (__bf16)wo[(c >> 6) * (DH * DM) + (c & 63) * DM + r];
}

__global__ __launch_bounds__(256) void prep_b_kernel(
    const float* __restrict__ bq, const float* __restrict__ bk,
    const float* __restrict__ bv, const float* __restrict__ bo,
    float* __restrict__ bqc, float* __restrict__ bkc,
    float* __restrict__ bvc, float* __restrict__ boc) {
  int i = blockIdx.x * 256 + threadIdx.x;   // over 1024
  bqc[i] = bq[i];   // [H][DH] flat == h*64+e ordering already
  bkc[i] = bk[i];
  bvc[i] = bv[i];
  float s = 0.f;
#pragma unroll
  for (int h = 0; h < H; ++h) s += bo[h * DM + i];
  boc[i] = s;
}

// ----------------------------- generic GEMM --------------------------------
// C[M][N] = A[M][K] * Bt[N][K]^T + bias[N]
// Block: 256 threads / 8 waves; tile 256(M) x 64(N); wave tile 32x64.
// B panel (64 rows x 32 k) staged in LDS by the TDM, double-buffered.
// MODE 0: bf16 row-major out; MODE 1: bf16 V-transposed out; MODE 2: f32 out.
template <int MODE>
__global__ __launch_bounds__(256) void gemm_kernel(
    const __bf16* __restrict__ A, const __bf16* __restrict__ Bt,
    const float* __restrict__ bias, void* __restrict__ Cout, int N, int K) {
  const int lane = threadIdx.x & 31;
  const int wave = threadIdx.x >> 5;
  const int m0 = blockIdx.x * 256 + wave * 32;
  const int n0 = blockIdx.y * 64;

  __shared__ __align__(16) __bf16 bsh[2][64 * 32];   // [buf][n_local*32 + k]

  auto stage = [&](int k, int buf) {
#if HAVE_TDM
    if (wave == 0)
      tdm_load_2d(lds_offset_of(&bsh[buf][0]), Bt + (size_t)n0 * K + k,
                  (unsigned)K, (unsigned)N, 32u, 64u, (unsigned)K);
#else
    const int t  = threadIdx.x;          // 256 threads, 8 halves each
    const int nl = t >> 2, kl = (t & 3) << 3;
    *(v8bf*)&bsh[buf][nl * 32 + kl] =
        *(const v8bf*)(Bt + (size_t)(n0 + nl) * K + k + kl);
#endif
  };

  v8f acc[2][4] = {};
  stage(0, 0);
  tensor_wait_barrier(wave == 0);

  for (int kk = 0; kk < K; kk += 32) {
    const int buf = (kk >> 5) & 1;
    if (kk + 32 < K) stage(kk + 32, buf ^ 1);
    __builtin_prefetch(A + (size_t)(m0 + (lane & 15)) * K + kk + 256, 0, 0);

    v16bf a0 = load_frag(A, m0,      K, kk, lane);
    v16bf a1 = load_frag(A, m0 + 16, K, kk, lane);
#pragma unroll
    for (int t = 0; t < 4; ++t) {
      v16bf b = load_frag(&bsh[buf][0], t * 16, 32, 0, lane);
      acc[0][t] = WMMA_BF16(a0, b, acc[0][t]);
      acc[1][t] = WMMA_BF16(a1, b, acc[1][t]);
    }
    tensor_wait_barrier(wave == 0);
  }

  const int hi8 = (lane >> 4) << 3;
#pragma unroll
  for (int f = 0; f < 2; ++f) {
#pragma unroll
    for (int t = 0; t < 4; ++t) {
      const int col = n0 + t * 16 + (lane & 15);
      const float bb = bias[col];
#pragma unroll
      for (int j = 0; j < 8; ++j) {
        const int row = m0 + f * 16 + j + hi8;
        const float val = acc[f][t][j] + bb;
        if (MODE == 0) {
          ((__bf16*)Cout)[(size_t)row * N + col] = (__bf16)val;
        } else if (MODE == 2) {
          ((float*)Cout)[(size_t)row * N + col] = val;
        } else {  // V transposed: Vt[(b*1024 + col)][s]
          const int bidx = row >> 11, s = row & (S - 1);
          ((__bf16*)Cout)[((size_t)(bidx * 1024 + col)) * S + s] = (__bf16)val;
        }
      }
    }
  }
}

// --------------------------- flash attention -------------------------------
// grid = (B*H, S/128); block = 256 (8 waves); each wave owns 16 query rows.
// K tile (32 keys x 64 dims) and V tile (64 dims x 32 keys) are workgroup-
// shared: staged into LDS by TDM (wave0 -> K, wave1 -> V), double-buffered.
__global__ __launch_bounds__(256) void attn_kernel(
    const __bf16* __restrict__ Qb, const __bf16* __restrict__ Kb,
    const __bf16* __restrict__ Vt, __bf16* __restrict__ Hd) {
  const int lane = threadIdx.x & 31;
  const int wave = threadIdx.x >> 5;
  const int bh = blockIdx.x;
  const int b = bh >> 4, h = bh & 15;
  const int q0 = blockIdx.y * 128 + wave * 16;

  const __bf16* Q  = Qb + (size_t)b * S * DM + h * DH;
  const __bf16* Kp = Kb + (size_t)b * S * DM + h * DH;
  const __bf16* Vp = Vt + (size_t)bh * DH * S;   // row e -> Vp[e*S + s]

  __shared__ __align__(16) __bf16 ksh[2][32 * 64];   // [buf][key_local*64 + e]
  __shared__ __align__(16) __bf16 vsh[2][64 * 32];   // [buf][e*32 + key_local]
  __shared__ __align__(16) __bf16 plds[8][16 * 32];  // per-wave P transpose
  __bf16* pl = plds[wave];

  auto stage = [&](int n0, int buf) {
#if HAVE_TDM
    if (wave == 0)
      tdm_load_2d(lds_offset_of(&ksh[buf][0]), Kp + (size_t)n0 * DM,
                  (unsigned)DH, (unsigned)S, 64u, 32u, (unsigned)DM);
    else if (wave == 1)
      tdm_load_2d(lds_offset_of(&vsh[buf][0]), Vp + n0,
                  (unsigned)S, (unsigned)DH, 32u, 64u, (unsigned)S);
#else
    const int t = threadIdx.x;
    {  // K: 32 rows x 8 chunks of 8 halves
      const int r = t >> 3, kc = (t & 7) << 3;
      *(v8bf*)&ksh[buf][r * 64 + kc] = *(const v8bf*)(Kp + (size_t)(n0 + r) * DM + kc);
    }
    {  // V: 64 rows x 4 chunks of 8 halves
      const int r = t >> 2, kc = (t & 3) << 3;
      *(v8bf*)&vsh[buf][r * 32 + kc] = *(const v8bf*)(Vp + (size_t)r * S + n0 + kc);
    }
#endif
  };

  // Q fragments persist across the key loop (16 rows x 64 dims).
  v16bf aq0 = load_frag(Q, q0, DM, 0, lane);
  v16bf aq1 = load_frag(Q, q0, DM, 32, lane);

  float mrow[8], lrow[8];
  v8f o[4] = {};
#pragma unroll
  for (int j = 0; j < 8; ++j) { mrow[j] = -1e30f; lrow[j] = 0.f; }
  const float scale = 0.125f;  // 1/sqrt(64)

  stage(0, 0);
  tensor_wait_barrier(wave < 2);

  for (int n0 = 0; n0 < S; n0 += 32) {
    const int buf = (n0 >> 5) & 1;
    if (n0 + 32 < S) stage(n0 + 32, buf ^ 1);

    // S = Q*K^T : two 16x16 score tiles, contraction 64 = 2 x WMMA-K32
    const __bf16* kt = &ksh[buf][0];
    v16bf k00 = load_frag(kt, 0,  64, 0,  lane);
    v16bf k01 = load_frag(kt, 0,  64, 32, lane);
    v16bf k10 = load_frag(kt, 16, 64, 0,  lane);
    v16bf k11 = load_frag(kt, 16, 64, 32, lane);
    v8f s0 = {}; s0 = WMMA_BF16(aq0, k00, s0); s0 = WMMA_BF16(aq1, k01, s0);
    v8f s1 = {}; s1 = WMMA_BF16(aq0, k10, s1); s1 = WMMA_BF16(aq1, k11, s1);

    // Online softmax.  C layout: VGPR j holds row (j + 8*(lane>=16)),
    // key column = lane&15 (+16 for second tile).  Row reductions span the
    // 16-lane half-group via xor shuffles (masks < 16 stay in-group).
    float corr[8];
#pragma unroll
    for (int j = 0; j < 8; ++j) {
      float a0 = s0[j] * scale, a1 = s1[j] * scale;
      float cm = fmaxf(a0, a1);
#pragma unroll
      for (int msk = 1; msk < 16; msk <<= 1)
        cm = fmaxf(cm, __shfl_xor(cm, msk, 32));
      float nm = fmaxf(mrow[j], cm);
      corr[j] = __expf(mrow[j] - nm);
      mrow[j] = nm;
      float p0 = __expf(a0 - nm), p1 = __expf(a1 - nm);
      float rs = p0 + p1;
#pragma unroll
      for (int msk = 1; msk < 16; msk <<= 1)
        rs += __shfl_xor(rs, msk, 32);
      lrow[j] = lrow[j] * corr[j] + rs;
      const int row = j + ((lane >> 4) << 3);
      pl[row * 32 + (lane & 15)]      = (__bf16)p0;
      pl[row * 32 + (lane & 15) + 16] = (__bf16)p1;
    }
#pragma unroll
    for (int t = 0; t < 4; ++t)
#pragma unroll
      for (int j = 0; j < 8; ++j) o[t][j] *= corr[j];

    __syncthreads();  // P transpose visible (per-wave region, hw keeps order)

    // O += P * V : P re-read as A fragment (16x32), V from LDS [e][key].
    v16bf ap = load_frag(pl, 0, 32, 0, lane);
#pragma unroll
    for (int t = 0; t < 4; ++t) {
      v16bf bv = load_frag(&vsh[buf][0], t * 16, 32, 0, lane);
      o[t] = WMMA_BF16(ap, bv, o[t]);
    }
    tensor_wait_barrier(wave < 2);
  }

  const int hi8 = (lane >> 4) << 3;
#pragma unroll
  for (int t = 0; t < 4; ++t) {
#pragma unroll
    for (int j = 0; j < 8; ++j) {
      const int row = q0 + j + hi8;
      const float val = o[t][j] / lrow[j];
      Hd[((size_t)b * S + row) * DM + h * DH + t * 16 + (lane & 15)] = (__bf16)val;
    }
  }
}

// ------------------------------- launcher ----------------------------------

extern "C" void kernel_launch(void* const* d_in, const int* in_sizes, int n_in,
                              void* d_out, int out_size, void* d_ws, size_t ws_size,
                              hipStream_t stream) {
  const float* x  = (const float*)d_in[0];
  const float* wq = (const float*)d_in[1];
  const float* bq = (const float*)d_in[2];
  const float* wk = (const float*)d_in[3];
  const float* bk = (const float*)d_in[4];
  const float* wv = (const float*)d_in[5];
  const float* bv = (const float*)d_in[6];
  const float* wo = (const float*)d_in[7];
  const float* bo = (const float*)d_in[8];

  char* w = (char*)d_ws;
  auto take = [&](size_t bytes) {
    void* p = (void*)w;
    w += (bytes + 255) & ~(size_t)255;
    return p;
  };
  __bf16* Xb  = (__bf16*)take((size_t)M * DM * 2);
  __bf16* Wqt = (__bf16*)take((size_t)DM * DM * 2);
  __bf16* Wkt = (__bf16*)take((size_t)DM * DM * 2);
  __bf16* Wvt = (__bf16*)take((size_t)DM * DM * 2);
  __bf16* Wot = (__bf16*)take((size_t)DM * DM * 2);
  float*  bqc = (float*)take(DM * 4);
  float*  bkc = (float*)take(DM * 4);
  float*  bvc = (float*)take(DM * 4);
  float*  boc = (float*)take(DM * 4);
  __bf16* Qb  = (__bf16*)take((size_t)M * DM * 2);
  __bf16* Kb  = (__bf16*)take((size_t)M * DM * 2);
  __bf16* Vt  = (__bf16*)take((size_t)BB * DM * S * 2);
  __bf16* Hd  = (__bf16*)take((size_t)M * DM * 2);

  // 1) convert inputs to bf16 / transposed layouts
  cvt_x_kernel<<<(M * DM + 255) / 256, 256, 0, stream>>>(x, Xb, M * DM);
  prep_w_kernel<<<(DM * DM) / 256, 256, 0, stream>>>(wq, wk, wv, wo, Wqt, Wkt, Wvt, Wot);
  prep_b_kernel<<<DM / 256, 256, 0, stream>>>(bq, bk, bv, bo, bqc, bkc, bvc, boc);

  // 2) QKV projections (WMMA GEMMs, 256x64 tiles, TDM-staged B panels)
  dim3 ggrid(M / 256, DM / 64);
  gemm_kernel<0><<<ggrid, 256, 0, stream>>>(Xb, Wqt, bqc, (void*)Qb, DM, DM);
  gemm_kernel<0><<<ggrid, 256, 0, stream>>>(Xb, Wkt, bkc, (void*)Kb, DM, DM);
  gemm_kernel<1><<<ggrid, 256, 0, stream>>>(Xb, Wvt, bvc, (void*)Vt, DM, DM);

  // 3) flash attention (TDM-staged K/V tiles)
  attn_kernel<<<dim3(BB * H, S / 128), 256, 0, stream>>>(Qb, Kb, Vt, Hd);

  // 4) output projection (fp32 result)
  gemm_kernel<2><<<ggrid, 256, 0, stream>>>(Hd, Wot, boc, d_out, DM, DM);
}
